// ISA2D_43911745634357
// MI455X (gfx1250) — matrix operand order repaired and checked by reading, not compiled, not run on hardware
//
#include <hip/hip_runtime.h>
#include <stdint.h>

// ---------------------------------------------------------------------------
// MI455X (gfx1250): all heavy math via v_wmma_f32_16x16x32_bf16; all LDS tile
// staging via global_load_async_to_lds_b128 (ASYNCcnt) for copy/compute overlap.
// Fragment layouts used in this file (C/D as in CDNA5 ISA):
//   A (16x32 bf16): lane L -> m=L%16, half=L/16; dword v -> K = 2v+16*half (same contiguous order as B)
//   B (32x16 bf16): lane L -> n=L%16, half=L/16; dword v -> K = 2v+16*half (contiguous; not the ISA table order)
//   C/D (16x16 f32): lane L -> n=L%16; vgpr v -> m = v + 8*(L/16)
// ---------------------------------------------------------------------------

typedef __attribute__((ext_vector_type(16))) __bf16 bf16x16;
typedef __attribute__((ext_vector_type(8)))  float  f32x8;

union FragBF { bf16x16 v; uint32_t u[8]; uint4 q[2]; };

__device__ __forceinline__ unsigned short f2bf(float f) {
  uint32_t x = __float_as_uint(f);
  x += 0x7fffu + ((x >> 16) & 1u);          // round-to-nearest-even
  return (unsigned short)(x >> 16);
}

__device__ __forceinline__ int a_k0(int v, int h) {
  return 2 * v + 16 * h;
}

// low 32 bits of a generic pointer to __shared__ == LDS byte offset
__device__ __forceinline__ uint32_t lds_off(const void* p) {
  return (uint32_t)(uintptr_t)p;
}

// Async DMA: 16 bytes global -> LDS, tracked by ASYNCcnt (no VGPR data).
__device__ __forceinline__ void async_b128(uint32_t lds, const void* gaddr) {
  asm volatile("global_load_async_to_lds_b128 %0, %1, off"
               :: "v"(lds), "v"(gaddr) : "memory");
}
__device__ __forceinline__ void wait_async0() {
  asm volatile("s_wait_asynccnt 0x0" ::: "memory");
}

#define REGUL 0.08838834764831845f  /* sqrt(2/256) */

// ---------------------------------------------------------------------------
// fp32 -> bf16 bulk convert (vectorized by 4)
// ---------------------------------------------------------------------------
__global__ __launch_bounds__(256) void f32_to_bf16_kernel(
    const float* __restrict__ in, unsigned short* __restrict__ out, int n4) {
  int i = blockIdx.x * 256 + threadIdx.x;
  if (i >= n4) return;
  float4 f = ((const float4*)in)[i];
  uint2 o;
  o.x = (uint32_t)f2bf(f.x) | ((uint32_t)f2bf(f.y) << 16);
  o.y = (uint32_t)f2bf(f.z) | ((uint32_t)f2bf(f.w) << 16);
  ((uint2*)out)[i] = o;
}

// ---------------------------------------------------------------------------
// BN (inference) -> per-channel scale/bias.  bn = [4 sets][gamma,beta,mean,var][128]
// ---------------------------------------------------------------------------
__global__ __launch_bounds__(256) void bn_prep_kernel(
    const float* __restrict__ bn, float* __restrict__ sc, float* __restrict__ bi) {
  int i = blockIdx.x * 256 + threadIdx.x;
  if (i >= 512) return;
  int set = i >> 7, c = i & 127;
  const float* base = bn + set * 512;
  float g = base[c], b = base[128 + c], m = base[256 + c], v = base[384 + c];
  float s = g * rsqrtf(v + 1e-3f);
  sc[i] = s;
  bi[i] = b - m * s;
}

// ---------------------------------------------------------------------------
// Pack fp32 weight [K][Ncols] into B-fragment dwords:
//   dst[((ntile*KB+kb)*32+lane)*8+v] = bf16pair(W[k0][n], W[k0+1][n])
//   n = ntile*16 + lane%16, k0 = kb*32 + 2v + 16*(lane/16)
// ---------------------------------------------------------------------------
__global__ __launch_bounds__(256) void pack_w_kernel(
    const float* __restrict__ W, uint32_t* __restrict__ dst, int K, int Ncols) {
  int tid = blockIdx.x * 256 + threadIdx.x;
  int KB = K >> 5;
  int total = KB * (Ncols >> 4) * 256;
  if (tid >= total) return;
  int v     = tid & 7;
  int lane  = (tid >> 3) & 31;
  int kb    = (tid >> 8) % KB;
  int ntile = (tid >> 8) / KB;
  int n  = ntile * 16 + (lane & 15);
  int k0 = kb * 32 + 2 * v + 16 * (lane >> 4);
  unsigned short lo = f2bf(W[(size_t)k0 * Ncols + n]);
  unsigned short hi = f2bf(W[(size_t)(k0 + 1) * Ncols + n]);
  dst[tid] = (uint32_t)lo | ((uint32_t)hi << 16);
}

// ---------------------------------------------------------------------------
// Tiled bf16 WMMA GEMM:  Out[M,Ncols] = A[M,K] @ W[K,Ncols]  (+BN+ReLU)
// Block: 256 threads (8 waves). Tile: 128 rows x 128 cols. K step 32.
// A tile and B weight chunk double-buffered in LDS via async-to-LDS DMA.
// ---------------------------------------------------------------------------
template <int KB, bool BN, bool F32OUT>
__global__ __launch_bounds__(256) void gemm_wmma_kernel(
    const unsigned short* __restrict__ A, const uint32_t* __restrict__ Wpk,
    void* __restrict__ Out, int Ncols,
    const float* __restrict__ bnscale, const float* __restrict__ bnbias) {
  __shared__ unsigned short Asm[2][128][32];  // 2 x 8KB
  __shared__ uint32_t Bsm[2][2048];           // 2 x 8KB (8 ntiles x 256 dwords)
  const int K = KB * 32;
  const int tid  = threadIdx.x;
  const int wave = tid >> 5, lane = tid & 31;
  const int n16  = lane & 15, half = lane >> 4;
  const int mbase = blockIdx.x * 128;
  const int cbase = blockIdx.y * 128;
  const int cb16  = cbase >> 4;

  f32x8 acc[8] = {};

  const int arow = tid >> 1, aseg = tid & 1;
  const char* abase =
      (const char*)(A + (size_t)(mbase + arow) * K + aseg * 16);

  auto stage = [&](int kb, int buf) {
    // A tile: 2 x b128 per thread (32B), coalesced rows
    const char* ga = abase + (size_t)kb * 64;
    async_b128(lds_off(&Asm[buf][arow][aseg * 16 + 0]), ga);
    async_b128(lds_off(&Asm[buf][arow][aseg * 16 + 8]), ga + 16);
    // B chunk for this kb: 8 tiles x 1KB; 2 x b128 per thread
#pragma unroll
    for (int r = 0; r < 2; ++r) {
      int i = tid + r * 256;              // 0..511
      int nt = i >> 6, rest = i & 63;
      const uint32_t* gb =
          Wpk + (size_t)(((cb16 + nt) * KB + kb) * 256 + rest * 4);
      async_b128(lds_off(&Bsm[buf][nt * 256 + rest * 4]), gb);
    }
  };

  stage(0, 0);
  wait_async0();
  __syncthreads();

  for (int kb = 0; kb < KB; ++kb) {
    if (kb + 1 < KB) stage(kb + 1, (kb + 1) & 1);
    const int buf = kb & 1;

    FragBF af;
    {
      const int row = wave * 16 + n16;
#pragma unroll
      for (int v = 0; v < 8; ++v)
        af.u[v] = *(const uint32_t*)&Asm[buf][row][a_k0(v, half)];
    }
#pragma unroll
    for (int nt = 0; nt < 8; ++nt) {
      const uint4* bp = (const uint4*)&Bsm[buf][(nt * 32 + lane) * 8];
      FragBF bf;
      bf.q[0] = bp[0];
      bf.q[1] = bp[1];
      acc[nt] = __builtin_amdgcn_wmma_f32_16x16x32_bf16(
          false, af.v, false, bf.v, (short)0, acc[nt], false, false);
    }
    wait_async0();
    __syncthreads();
  }

#pragma unroll
  for (int nt = 0; nt < 8; ++nt) {
    const int col = cbase + nt * 16 + n16;
    float sc = 1.f, bi = 0.f;
    if (BN) { sc = bnscale[col]; bi = bnbias[col]; }
#pragma unroll
    for (int v = 0; v < 8; ++v) {
      int row = mbase + wave * 16 + v + 8 * half;
      float y = acc[nt][v];
      if (BN) y = fmaxf(y * sc + bi, 0.f);
      if (F32OUT)
        ((float*)Out)[(size_t)row * Ncols + col] = y;
      else
        ((unsigned short*)Out)[(size_t)row * Ncols + col] = f2bf(y);
    }
  }
}

// ---------------------------------------------------------------------------
// Stage-1 attention: one block (4 waves) per sequence of L=32
// s = ((b*8+ph)*8+pw)*32+qh ; pixel(j) = (b*256+qh*8+ph)*256 + j*8 + pw
// ---------------------------------------------------------------------------
__global__ __launch_bounds__(128) void attn_stage1_kernel(
    const unsigned short* __restrict__ TH, const unsigned short* __restrict__ PHI,
    const unsigned short* __restrict__ G, unsigned short* __restrict__ O) {
  __shared__ unsigned short thS[32][128];
  __shared__ unsigned short phS[32][128];
  __shared__ uint32_t gpS[16][128];   // G pair-packed: [k/2][col] = (G[k],G[k+1])
  __shared__ float S[32][33];
  __shared__ unsigned short Abf[32][32];
  __shared__ int pmap[32];

  const int tid = threadIdx.x;
  const int wave = tid >> 5, lane = tid & 31;
  const int n16 = lane & 15, half = lane >> 4;
  const int s = blockIdx.x;

  if (tid < 32) {
    int qh = s & 31, pw = (s >> 5) & 7, ph = (s >> 8) & 7, b = s >> 11;
    pmap[tid] = (b * 256 + qh * 8 + ph) * 256 + tid * 8 + pw;
  }
  __syncthreads();

  // theta/phi tiles via async DMA to LDS
  for (int i = tid; i < 512; i += 128) {
    int j = i >> 4, seg = i & 15;
    size_t src = (size_t)pmap[j] * 128 + seg * 8;
    async_b128(lds_off(&thS[j][seg * 8]), TH + src);
    async_b128(lds_off(&phS[j][seg * 8]), PHI + src);
  }
  // G: load + pair-pack in registers
  unsigned short* gp16 = (unsigned short*)&gpS[0][0];
  for (int i = tid; i < 512; i += 128) {
    int k = i >> 4, seg = i & 15;
    uint4 q = *(const uint4*)(G + (size_t)pmap[k] * 128 + seg * 8);
    const unsigned short* e = (const unsigned short*)&q;
#pragma unroll
    for (int c = 0; c < 8; ++c)
      gp16[(((k >> 1) * 128) + seg * 8 + c) * 2 + (k & 1)] = e[c];
  }
  wait_async0();
  __syncthreads();

  {  // scores: wave (qi,kj) owns one 16x16 tile, K=128
    const int qi = wave & 1, kj = wave >> 1;
    f32x8 acc = {};
#pragma unroll
    for (int kc = 0; kc < 4; ++kc) {
      FragBF a, b;
#pragma unroll
      for (int v = 0; v < 8; ++v) {
        a.u[v] = *(const uint32_t*)&thS[qi * 16 + n16][kc * 32 + a_k0(v, half)];
        b.u[v] = *(const uint32_t*)&phS[kj * 16 + n16][kc * 32 + 2 * v + 16 * half];
      }
      acc = __builtin_amdgcn_wmma_f32_16x16x32_bf16(false, a.v, false, b.v,
                                                    (short)0, acc, false, false);
    }
#pragma unroll
    for (int v = 0; v < 8; ++v)
      S[qi * 16 + v + 8 * half][kj * 16 + n16] = acc[v] * REGUL;
  }
  __syncthreads();

  if (tid < 32) {  // row softmax
    float mx = -3e38f;
#pragma unroll
    for (int j = 0; j < 32; ++j) mx = fmaxf(mx, S[tid][j]);
    float sum = 0.f;
#pragma unroll
    for (int j = 0; j < 32; ++j) {
      float e = __expf(S[tid][j] - mx);
      S[tid][j] = e;
      sum += e;
    }
    float inv = 1.f / sum;
#pragma unroll
    for (int j = 0; j < 32; ++j) Abf[tid][j] = f2bf(S[tid][j] * inv);
  }
  __syncthreads();

  {  // out = A(32x32) @ G(32x128): wave -> rows (w&1)*16, cols (w>>1)*64
    const int r = (wave & 1) * 16;
    const int cb = (wave >> 1) * 64;
    FragBF a;
#pragma unroll
    for (int v = 0; v < 8; ++v)
      a.u[v] = *(const uint32_t*)&Abf[r + n16][a_k0(v, half)];
#pragma unroll
    for (int ct = 0; ct < 4; ++ct) {
      FragBF b;
#pragma unroll
      for (int v = 0; v < 8; ++v)
        b.u[v] = gpS[v + 8 * half][cb + ct * 16 + n16];
      f32x8 o = {};
      o = __builtin_amdgcn_wmma_f32_16x16x32_bf16(false, a.v, false, b.v,
                                                  (short)0, o, false, false);
#pragma unroll
      for (int v = 0; v < 8; ++v)
        O[(size_t)pmap[r + v + 8 * half] * 128 + cb + ct * 16 + n16] =
            f2bf(o[v]);
    }
  }
}

// ---------------------------------------------------------------------------
// Stage-2 attention: L=8.  Two sequences batched per 16x16 tile; cross blocks
// masked to -inf (softmax weight exactly 0, so A@Gcat stays correct).
// s = ((b*32+qh)*32+qw)*8+ph ; pixel(j)=(b*256+qh*8+ph)*256+qw*8+j
// out row scatter: fr(s,j) = (b*256+ph*32+qh)*256 + j*32 + qw
// ---------------------------------------------------------------------------
__device__ __forceinline__ int s2_pbase(int s) {
  int ph = s & 7, qw = (s >> 3) & 31, qh = (s >> 8) & 31, b = s >> 13;
  return (b * 256 + qh * 8 + ph) * 256 + qw * 8;
}
__device__ __forceinline__ int s2_frow(int s, int j) {
  int ph = s & 7, qw = (s >> 3) & 31, qh = (s >> 8) & 31, b = s >> 13;
  return (b * 256 + ph * 32 + qh) * 256 + j * 32 + qw;
}

__global__ __launch_bounds__(128) void attn_stage2_kernel(
    const unsigned short* __restrict__ TH, const unsigned short* __restrict__ PHI,
    const unsigned short* __restrict__ G, unsigned short* __restrict__ O) {
  __shared__ unsigned short thS[4][16][128];
  __shared__ unsigned short phS[4][16][128];
  __shared__ uint32_t gpS[4][8][128];
  __shared__ unsigned short Abf[4][16][32];

  const int tid = threadIdx.x, wave = tid >> 5, lane = tid & 31;
  const int n16 = lane & 15, half = lane >> 4;
  const int pair = blockIdx.x * 4 + wave;
  const int s0 = pair * 2, s1 = s0 + 1;
  const int pb0 = s2_pbase(s0), pb1 = s2_pbase(s1);

  for (int i = lane; i < 256; i += 32) {
    int r = i >> 4, seg = i & 15;
    int p = (r < 8) ? (pb0 + r) : (pb1 + (r - 8));
    async_b128(lds_off(&thS[wave][r][seg * 8]), TH + (size_t)p * 128 + seg * 8);
    async_b128(lds_off(&phS[wave][r][seg * 8]), PHI + (size_t)p * 128 + seg * 8);
  }
  unsigned short* gp16 = (unsigned short*)&gpS[wave][0][0];
  for (int i = lane; i < 256; i += 32) {
    int k = i >> 4, seg = i & 15;
    int p = (k < 8) ? (pb0 + k) : (pb1 + (k - 8));
    uint4 q = *(const uint4*)(G + (size_t)p * 128 + seg * 8);
    const unsigned short* e = (const unsigned short*)&q;
#pragma unroll
    for (int c = 0; c < 8; ++c)
      gp16[(((k >> 1) * 128) + seg * 8 + c) * 2 + (k & 1)] = e[c];
  }
  wait_async0();
  __syncthreads();

  // scores (one 16x16 tile per wave, K=128)
  f32x8 acc = {};
#pragma unroll
  for (int kc = 0; kc < 4; ++kc) {
    FragBF a, b;
#pragma unroll
    for (int v = 0; v < 8; ++v) {
      a.u[v] = *(const uint32_t*)&thS[wave][n16][kc * 32 + a_k0(v, half)];
      b.u[v] = *(const uint32_t*)&phS[wave][n16][kc * 32 + 2 * v + 16 * half];
    }
    acc = __builtin_amdgcn_wmma_f32_16x16x32_bf16(false, a.v, false, b.v,
                                                  (short)0, acc, false, false);
  }

  // masked softmax: row m=v+8*half lives in 8 valid lanes of one half
  const bool valid = ((n16 < 8) == (half == 0));
#pragma unroll
  for (int v = 0; v < 8; ++v) {
    float x = valid ? acc[v] * REGUL : -3.0e38f;
    float m = x;
    m = fmaxf(m, __shfl_xor(m, 1));
    m = fmaxf(m, __shfl_xor(m, 2));
    m = fmaxf(m, __shfl_xor(m, 4));
    float e = valid ? __expf(x - m) : 0.f;
    float su = e;
    su += __shfl_xor(su, 1);
    su += __shfl_xor(su, 2);
    su += __shfl_xor(su, 4);
    float aw = valid ? (e / su) : 0.f;
    Abf[wave][v + 8 * half][n16] = f2bf(aw);
    Abf[wave][v + 8 * half][n16 + 16] = 0;  // K pad 16..31
  }
  __syncthreads();

  // out = A(16x16 pad to K=32) @ Gcat(16x128 pad to 32)
  FragBF a;
#pragma unroll
  for (int v = 0; v < 8; ++v)
    a.u[v] = *(const uint32_t*)&Abf[wave][n16][a_k0(v, half)];
#pragma unroll
  for (int ct = 0; ct < 8; ++ct) {
    FragBF b;
#pragma unroll
    for (int v = 0; v < 8; ++v)
      b.u[v] = (half == 0) ? gpS[wave][v][ct * 16 + n16] : 0u;  // K>=16 is pad
    f32x8 o = {};
    o = __builtin_amdgcn_wmma_f32_16x16x32_bf16(false, a.v, false, b.v,
                                                (short)0, o, false, false);
#pragma unroll
    for (int v = 0; v < 8; ++v) {
      int m = v + 8 * half;
      int sX = (m < 8) ? s0 : s1;
      O[(size_t)s2_frow(sX, m & 7) * 128 + ct * 16 + n16] = f2bf(o[v]);
    }
  }
}

// ---------------------------------------------------------------------------
// Host side
// ---------------------------------------------------------------------------
extern "C" void kernel_launch(void* const* d_in, const int* in_sizes, int n_in,
                              void* d_out, int out_size, void* d_ws, size_t ws_size,
                              hipStream_t stream) {
  (void)in_sizes; (void)n_in; (void)out_size; (void)ws_size;
  const int N = 4 * 256 * 256;  // pixels

  const float* x      = (const float*)d_in[0];
  const float* a1_in  = (const float*)d_in[1];  // [3][256][128]
  const float* a1_mid = (const float*)d_in[2];  // [2][128][128]
  const float* a1_rho = (const float*)d_in[3];  // [128][256]
  const float* a1_bn  = (const float*)d_in[4];  // [4][4][128]
  const float* a2_in  = (const float*)d_in[5];
  const float* a2_mid = (const float*)d_in[6];
  const float* a2_rho = (const float*)d_in[7];
  const float* a2_bn  = (const float*)d_in[8];

  size_t off = 0;
  auto alloc = [&](size_t bytes) -> char* {
    char* p = (char*)d_ws + off;
    off = (off + bytes + 255) & ~(size_t)255;
    return p;
  };
  const size_t NB16 = (size_t)N * 128 * 2;
  const size_t NB32 = (size_t)N * 256 * 2;

  unsigned short* Xb  = (unsigned short*)alloc(NB32);
  unsigned short* Y1  = (unsigned short*)alloc(NB32);
  unsigned short* T1  = (unsigned short*)alloc(NB16);
  unsigned short* P1  = (unsigned short*)alloc(NB16);
  unsigned short* Gb  = (unsigned short*)alloc(NB16);
  unsigned short* THb = (unsigned short*)alloc(NB16);
  unsigned short* PHb = (unsigned short*)alloc(NB16);
  unsigned short* Ob  = (unsigned short*)alloc(NB16);

  // packed weights: per stage {in0,in1,in2 (64KB), mid0,mid1 (32KB), rho (64KB)}
  uint32_t* pk[2][6];
  for (int st = 0; st < 2; ++st) {
    pk[st][0] = (uint32_t*)alloc(16384 * 4);
    pk[st][1] = (uint32_t*)alloc(16384 * 4);
    pk[st][2] = (uint32_t*)alloc(16384 * 4);
    pk[st][3] = (uint32_t*)alloc(8192 * 4);
    pk[st][4] = (uint32_t*)alloc(8192 * 4);
    pk[st][5] = (uint32_t*)alloc(16384 * 4);
  }
  float* sc = (float*)alloc(2 * 512 * 4);  // [stage][set][128]
  float* bi = (float*)alloc(2 * 512 * 4);

  // 1) convert activations to bf16
  {
    int n4 = N * 256 / 4;
    f32_to_bf16_kernel<<<(n4 + 255) / 256, 256, 0, stream>>>(x, Xb, n4);
  }
  // 2) BN prep
  bn_prep_kernel<<<2, 256, 0, stream>>>(a1_bn, sc, bi);
  bn_prep_kernel<<<2, 256, 0, stream>>>(a2_bn, sc + 512, bi + 512);

  // 3) pack weights into B-fragment layout
  auto pack = [&](const float* W, uint32_t* dst, int K, int Ncols) {
    int total = (K >> 5) * (Ncols >> 4) * 256;
    pack_w_kernel<<<(total + 255) / 256, 256, 0, stream>>>(W, dst, K, Ncols);
  };
  const float* win[2]  = {a1_in, a2_in};
  const float* wmid[2] = {a1_mid, a2_mid};
  const float* wrho[2] = {a1_rho, a2_rho};
  for (int st = 0; st < 2; ++st) {
    pack(win[st] + 0 * 256 * 128, pk[st][0], 256, 128);
    pack(win[st] + 1 * 256 * 128, pk[st][1], 256, 128);
    pack(win[st] + 2 * 256 * 128, pk[st][2], 256, 128);
    pack(wmid[st] + 0 * 128 * 128, pk[st][3], 128, 128);
    pack(wmid[st] + 1 * 128 * 128, pk[st][4], 128, 128);
    pack(wrho[st], pk[st][5], 128, 256);
  }

  const dim3 g1(N / 128, 1), g2(N / 128, 2), blk(256);

  for (int st = 0; st < 2; ++st) {
    const unsigned short* A = (st == 0) ? Xb : Y1;
    const float* scs = sc + st * 512;
    const float* bis = bi + st * 512;

    // theta/phi/gamma input convs (K=256 -> KB=8)
    gemm_wmma_kernel<8, true, false><<<g1, blk, 0, stream>>>(
        A, pk[st][0], T1, 128, scs + 0 * 128, bis + 0 * 128);   // bn[0]
    gemm_wmma_kernel<8, true, false><<<g1, blk, 0, stream>>>(
        A, pk[st][1], P1, 128, scs + 2 * 128, bis + 2 * 128);   // bn[2]
    gemm_wmma_kernel<8, false, false><<<g1, blk, 0, stream>>>(
        A, pk[st][2], Gb, 128, nullptr, nullptr);               // gamma
    // mid convs (K=128 -> KB=4)
    gemm_wmma_kernel<4, true, false><<<g1, blk, 0, stream>>>(
        T1, pk[st][3], THb, 128, scs + 1 * 128, bis + 1 * 128); // bn[1]
    gemm_wmma_kernel<4, true, false><<<g1, blk, 0, stream>>>(
        P1, pk[st][4], PHb, 128, scs + 3 * 128, bis + 3 * 128); // bn[3]
    // attention
    if (st == 0)
      attn_stage1_kernel<<<8192, 128, 0, stream>>>(THb, PHb, Gb, Ob);
    else
      attn_stage2_kernel<<<4096, 128, 0, stream>>>(THb, PHb, Gb, Ob);
    // rho conv (K=128 -> KB=4)
    if (st == 0)
      gemm_wmma_kernel<4, false, false><<<g2, blk, 0, stream>>>(
          Ob, pk[st][5], Y1, 256, nullptr, nullptr);
    else
      gemm_wmma_kernel<4, false, true><<<g2, blk, 0, stream>>>(
          Ob, pk[st][5], (float*)d_out, 256, nullptr, nullptr);
  }
}
